// GraphConvLayer_28286654611817
// MI455X (gfx1250) — matrix operand order repaired and checked by reading, not compiled, not run on hardware
//
#include <hip/hip_runtime.h>
#include <hip/hip_bf16.h>
#include <math.h>

// ---------------------------------------------------------------------------
// CDNA5 (gfx1250) WMMA types
// ---------------------------------------------------------------------------
typedef _Float16 f16x8  __attribute__((ext_vector_type(8)));
typedef _Float16 f16x16 __attribute__((ext_vector_type(16)));
typedef float    f32x8  __attribute__((ext_vector_type(8)));

// Exact-erf GELU (torch default)
__device__ __forceinline__ float gelu_exact(float x) {
    return 0.5f * x * (1.0f + erff(x * 0.70710678118654752f));
}

// ---------------------------------------------------------------------------
// Fragment loaders matching ISA 7.12.2 16-bit layouts (wave32).
// A (16xK row-major in LDS): lane&15 = row M; lanes 0-15 hold K {0..7,16..23},
// lanes 16-31 hold K {8..15,24..31} of each 32-wide chunk.
// B is fed from a TRANSPOSED weight buffer Wt[n][k]; lane&15 = column N,
// lanes 0-15 hold K {0..7,16..23}, lanes 16-31 hold K {8..15,24..31} (same as A).
// ---------------------------------------------------------------------------
__device__ __forceinline__ f16x16 frag_a(const _Float16* base, int stride,
                                         int k0, int lane) {
    const int row = lane & 15;
    const int off = ((lane >> 4) & 1) * 8;
    const _Float16* p = base + row * stride + k0 + off;
    f16x8 lo = *(const f16x8*)(p);
    f16x8 hi = *(const f16x8*)(p + 16);
    f16x16 r;
#pragma unroll
    for (int i = 0; i < 8; ++i) { r[i] = lo[i]; r[i + 8] = hi[i]; }
    return r;
}

__device__ __forceinline__ f16x16 frag_b(const _Float16* wt, int stride,
                                         int k0, int n0, int lane) {
    const int row = n0 + (lane & 15);
    const int kb  = k0 + ((lane >> 4) & 1) * 8;
    const _Float16* p = wt + row * stride + kb;
    f16x8 lo = *(const f16x8*)(p);
    f16x8 hi = *(const f16x8*)(p + 16);
    f16x16 r;
#pragma unroll
    for (int i = 0; i < 8; ++i) { r[i] = lo[i]; r[i + 8] = hi[i]; }
    return r;
}

__device__ __forceinline__ f32x8 wmma_f16(f16x16 a, f16x16 b, f32x8 c) {
    return __builtin_amdgcn_wmma_f32_16x16x32_f16(false, a, false, b,
                                                  (short)0, c, false, false);
}

// ---------------------------------------------------------------------------
// Kernel 0: init scratch (agg=0, denom=0, last_edge=-1)
// ---------------------------------------------------------------------------
__global__ void gc_init(float* agg, int* last_edge, float* denom, int N) {
    unsigned t = blockIdx.x * blockDim.x + threadIdx.x;
    if (t < (unsigned)N * 64u) agg[t] = 0.0f;
    if (t < (unsigned)N)       last_edge[t] = -1;
    if (t < 8u)                denom[t] = 0.0f;
}

// ---------------------------------------------------------------------------
// Kernel 1: per-node q projection  qn = x @ q_w + q_b   [N,64]
// 128 threads = 4 waves, each wave owns a 16-node tile.
// ---------------------------------------------------------------------------
__global__ __launch_bounds__(128) void gc_qn(const float* __restrict__ xf,
                                             const float* __restrict__ qw,
                                             const float* __restrict__ qb,
                                             float* __restrict__ qn, int N) {
    __shared__ alignas(16) _Float16 wbuf[64 * 64];      // q_w^T  (8 KB)
    __shared__ alignas(16) _Float16 abuf[4][16 * 64];   // 4x2 KB
    const int tid = threadIdx.x, lane = tid & 31, wave = tid >> 5;
    const int nbase = (blockIdx.x * 4 + wave) * 16;
    _Float16* A = abuf[wave];

    for (int idx = lane; idx < 16 * 64; idx += 32) {
        int i = idx >> 6, c = idx & 63;
        int n = nbase + i; if (n >= N) n = N - 1;
        A[i * 64 + c] = (_Float16)xf[(unsigned)n * 64u + c];
    }
    for (int i = tid; i < 64 * 64; i += 128) {
        int k = i >> 6, n = i & 63;
        wbuf[n * 64 + k] = (_Float16)qw[i];
    }
    __syncthreads();

    const int colb = lane & 15, rowb = ((lane >> 4) & 1) * 8;
    f16x16 af[2];
    af[0] = frag_a(A, 64, 0, lane);
    af[1] = frag_a(A, 64, 32, lane);
    f32x8 acc[4];
#pragma unroll
    for (int t = 0; t < 4; ++t) {
        float bias = qb[t * 16 + colb];
#pragma unroll
        for (int r = 0; r < 8; ++r) acc[t][r] = bias;
    }
    // k outer, tile inner: 4 independent accumulators -> hazard slots filled
#pragma unroll
    for (int kc = 0; kc < 2; ++kc)
#pragma unroll
        for (int t = 0; t < 4; ++t)
            acc[t] = wmma_f16(af[kc], frag_b(wbuf, 64, kc * 32, t * 16, lane), acc[t]);

    float* qrow = qn + (unsigned)nbase * 64u;
    if (nbase + 16 <= N) {
#pragma unroll
        for (int t = 0; t < 4; ++t)
#pragma unroll
            for (int r = 0; r < 8; ++r)
                qrow[(unsigned)(rowb + r) * 64u + t * 16 + colb] = acc[t][r];
    } else {
#pragma unroll
        for (int t = 0; t < 4; ++t)
#pragma unroll
            for (int r = 0; r < 8; ++r)
                if (nbase + rowb + r < N)
                    qrow[(unsigned)(rowb + r) * 64u + t * 16 + colb] = acc[t][r];
    }
}

// ---------------------------------------------------------------------------
// Kernel 2: edge-parallel message MLP + kv projection + attention logits.
// 128 threads = 4 waves, each wave owns a 16-edge tile (block = 64 edges).
// LDS: 24 KB shared phased weight buffer + per-wave A1 (16x192) / A2 (16x64).
// ---------------------------------------------------------------------------
__global__ __launch_bounds__(128) void gc_edge(
    const float* __restrict__ xf, const int* __restrict__ ei,
    const float* __restrict__ ea,
    const float* __restrict__ w1, const float* __restrict__ b1,
    const float* __restrict__ w2, const float* __restrict__ b2,
    const float* __restrict__ kvw, const float* __restrict__ kvb,
    const float* __restrict__ qn,
    float* __restrict__ m_out, float* __restrict__ attn_out,
    int* __restrict__ last_edge, int N, int E) {

    __shared__ alignas(16) _Float16 wbuf[64 * 192];     // 24 KB, reused 3x
    __shared__ alignas(16) _Float16 a1[4][16 * 192];    // 24 KB
    __shared__ alignas(16) _Float16 a2[4][16 * 64];     //  8 KB

    const int tid = threadIdx.x, lane = tid & 31, wave = tid >> 5;
    const int ebase = (blockIdx.x * 4 + wave) * 16;
    _Float16* A1 = a1[wave];
    _Float16* A2 = a2[wave];
    const int colb = lane & 15, rowb = ((lane >> 4) & 1) * 8;

    // speculative prefetch of edge indices 1024 edges (16 blocks) ahead (global_prefetch_b8)
    if (ebase + 1024 < E) __builtin_prefetch(ei + ebase + 1024, 0, 1);

    // -------- gather A1 = [s | d | edge_attr] as f16 --------
    for (int idx = lane; idx < 16 * 64; idx += 32) {
        int i = idx >> 6, c = idx & 63;
        int e = ebase + i; if (e >= E) e = E - 1;
        int s = ei[e], d = ei[E + e];
        A1[i * 192 + c]        = (_Float16)xf[(unsigned)s * 64u + c];
        A1[i * 192 + 64 + c]   = (_Float16)xf[(unsigned)d * 64u + c];
        A1[i * 192 + 128 + c]  = (_Float16)ea[(unsigned)e * 64u + c];
    }
    // -------- phase 1: msg layer 1 (192 -> 64) --------
    for (int i = tid; i < 192 * 64; i += 128) {
        int k = i >> 6, n = i & 63;
        wbuf[n * 192 + k] = (_Float16)w1[i];
    }
    __syncthreads();

    f16x16 af[6];
#pragma unroll
    for (int kc = 0; kc < 6; ++kc) af[kc] = frag_a(A1, 192, kc * 32, lane);

    f32x8 acc[4];
#pragma unroll
    for (int t = 0; t < 4; ++t) {
        float bias = b1[t * 16 + colb];
#pragma unroll
        for (int r = 0; r < 8; ++r) acc[t][r] = bias;
    }
#pragma unroll
    for (int kc = 0; kc < 6; ++kc)
#pragma unroll
        for (int t = 0; t < 4; ++t)
            acc[t] = wmma_f16(af[kc], frag_b(wbuf, 192, kc * 32, t * 16, lane), acc[t]);
#pragma unroll
    for (int t = 0; t < 4; ++t)
#pragma unroll
        for (int r = 0; r < 8; ++r)
            A2[(rowb + r) * 64 + t * 16 + colb] = (_Float16)gelu_exact(acc[t][r]);
    __syncthreads();

    // -------- phase 2: msg layer 2 (64 -> 64) -> m_out --------
    for (int i = tid; i < 64 * 64; i += 128) {
        int k = i >> 6, n = i & 63;
        wbuf[n * 64 + k] = (_Float16)w2[i];
    }
    __syncthreads();
    f16x16 a2f[2];
    a2f[0] = frag_a(A2, 64, 0, lane);
    a2f[1] = frag_a(A2, 64, 32, lane);
#pragma unroll
    for (int t = 0; t < 4; ++t) {
        float bias = b2[t * 16 + colb];
#pragma unroll
        for (int r = 0; r < 8; ++r) acc[t][r] = bias;
    }
#pragma unroll
    for (int kc = 0; kc < 2; ++kc)
#pragma unroll
        for (int t = 0; t < 4; ++t)
            acc[t] = wmma_f16(a2f[kc], frag_b(wbuf, 64, kc * 32, t * 16, lane), acc[t]);

    {   // single tile-level guard; 32-bit saddr+voffset addressing
        float* mrow = m_out + (unsigned)ebase * 64u;
        if (ebase + 16 <= E) {
#pragma unroll
            for (int t = 0; t < 4; ++t)
#pragma unroll
                for (int r = 0; r < 8; ++r)
                    mrow[(unsigned)(rowb + r) * 64u + t * 16 + colb] = acc[t][r];
        } else {
#pragma unroll
            for (int t = 0; t < 4; ++t)
#pragma unroll
                for (int r = 0; r < 8; ++r)
                    if (ebase + rowb + r < E)
                        mrow[(unsigned)(rowb + r) * 64u + t * 16 + colb] = acc[t][r];
        }
    }
    __syncthreads();

    // -------- phase 3: kv projection ([s|ea] 128 -> 64) --------
    for (int i = tid; i < 128 * 64; i += 128) {
        int k = i >> 6, n = i & 63;
        wbuf[n * 128 + k] = (_Float16)kvw[i];
    }
    __syncthreads();
    const int achunk[4] = {0, 1, 4, 5};        // A1 K-chunks for [s | ea]
    float* kvf = (float*)A1;                   // reuse own A1 region for f32 kv
#pragma unroll
    for (int t = 0; t < 4; ++t) {
        float bias = kvb[t * 16 + colb];
#pragma unroll
        for (int r = 0; r < 8; ++r) acc[t][r] = bias;
    }
#pragma unroll
    for (int kc = 0; kc < 4; ++kc)
#pragma unroll
        for (int t = 0; t < 4; ++t)
            acc[t] = wmma_f16(af[achunk[kc]],
                              frag_b(wbuf, 128, kc * 32, t * 16, lane), acc[t]);
#pragma unroll
    for (int t = 0; t < 4; ++t)
#pragma unroll
        for (int r = 0; r < 8; ++r)
            kvf[(rowb + r) * 64 + t * 16 + colb] = acc[t][r];

    // -------- attention logits: attn[e,h] = (q[dst,h,:] . kv[e,h,:]) / sqrt(8)
    for (int item = lane; item < 16 * 8; item += 32) {
        int i = item >> 3, h = item & 7;
        int e = ebase + i;
        if (e < E) {
            int d = ei[E + e];
            float s = 0.0f;
#pragma unroll
            for (int j = 0; j < 8; ++j)
                s += qn[(unsigned)d * 64u + h * 8 + j] * kvf[i * 64 + h * 8 + j];
            s *= 0.35355339059327373f;
            attn_out[(unsigned)e * 8u + h] = s;
            if (h == 0) atomicMax(&last_edge[d], e);   // "last write" per dst
        }
    }
}

// ---------------------------------------------------------------------------
// Kernel 3: softmax over the node axis (global per-head denominator).
// p[n,h] = exp(attn of last edge into n) or 0; denom[h] = sum_n p[n,h].
// ---------------------------------------------------------------------------
__global__ void gc_prep(const float* __restrict__ attn,
                        const int* __restrict__ last_edge,
                        float* __restrict__ p, float* __restrict__ denom,
                        int N) {
    __shared__ float part[8];
    if (threadIdx.x < 8) part[threadIdx.x] = 0.0f;
    __syncthreads();
    unsigned t = blockIdx.x * blockDim.x + threadIdx.x;
    if (t < (unsigned)N * 8u) {
        unsigned n = t >> 3, h = t & 7;
        int le = last_edge[n];
        float v = (le >= 0) ? __expf(attn[(unsigned)le * 8u + h]) : 0.0f;
        p[t] = v;
        atomicAdd(&part[h], v);
    }
    __syncthreads();
    if (threadIdx.x < 8) atomicAdd(&denom[threadIdx.x], part[threadIdx.x]);
}

// ---------------------------------------------------------------------------
// Kernel 4: weight messages and scatter-add to destination nodes.
// One thread per (edge, head): 8 columns each.
// ---------------------------------------------------------------------------
__global__ void gc_scatter(const float* __restrict__ m,
                           const int* __restrict__ ei,
                           const float* __restrict__ p,
                           const float* __restrict__ denom,
                           float* __restrict__ agg, int E) {
    unsigned t = blockIdx.x * blockDim.x + threadIdx.x;
    if (t >= (unsigned)E * 8u) return;
    unsigned e = t >> 3;
    int h = (int)(t & 7);
    int d = ei[E + e];
    float w = p[(unsigned)d * 8u + h] / denom[h];
#pragma unroll
    for (int j = 0; j < 8; ++j)
        atomicAdd(&agg[(unsigned)d * 64u + h * 8 + j],
                  m[e * 64u + h * 8 + j] * w);
}

// ---------------------------------------------------------------------------
// Kernel 5: output MLP + residual + layernorm.
// 128 threads = 4 waves, each wave owns a 16-node tile.
// ---------------------------------------------------------------------------
__global__ __launch_bounds__(128) void gc_out(
    const float* __restrict__ xf, const float* __restrict__ agg,
    const float* __restrict__ w1, const float* __restrict__ b1,
    const float* __restrict__ w2, const float* __restrict__ b2,
    const float* __restrict__ lng, const float* __restrict__ lnb,
    float* __restrict__ out, int N) {

    __shared__ alignas(16) _Float16 wbuf[64 * 128];     // 16 KB, reused 2x
    __shared__ alignas(16) _Float16 a1[4][16 * 128];    // 16 KB
    __shared__ alignas(16) _Float16 a2[4][16 * 64];     //  8 KB

    const int tid = threadIdx.x, lane = tid & 31, wave = tid >> 5;
    const int nbase = (blockIdx.x * 4 + wave) * 16;
    _Float16* A1 = a1[wave];
    _Float16* A2 = a2[wave];
    const int colb = lane & 15, rowb = ((lane >> 4) & 1) * 8;

    // gather [x | agg]
    for (int idx = lane; idx < 16 * 64; idx += 32) {
        int i = idx >> 6, c = idx & 63;
        int n = nbase + i; if (n >= N) n = N - 1;
        A1[i * 128 + c]      = (_Float16)xf[(unsigned)n * 64u + c];
        A1[i * 128 + 64 + c] = (_Float16)agg[(unsigned)n * 64u + c];
    }
    for (int i = tid; i < 128 * 64; i += 128) {
        int k = i >> 6, n = i & 63;
        wbuf[n * 128 + k] = (_Float16)w1[i];
    }
    __syncthreads();

    f16x16 af[4];
#pragma unroll
    for (int kc = 0; kc < 4; ++kc) af[kc] = frag_a(A1, 128, kc * 32, lane);
    f32x8 acc[4];
#pragma unroll
    for (int t = 0; t < 4; ++t) {
        float bias = b1[t * 16 + colb];
#pragma unroll
        for (int r = 0; r < 8; ++r) acc[t][r] = bias;
    }
#pragma unroll
    for (int kc = 0; kc < 4; ++kc)
#pragma unroll
        for (int t = 0; t < 4; ++t)
            acc[t] = wmma_f16(af[kc], frag_b(wbuf, 128, kc * 32, t * 16, lane), acc[t]);
#pragma unroll
    for (int t = 0; t < 4; ++t)
#pragma unroll
        for (int r = 0; r < 8; ++r)
            A2[(rowb + r) * 64 + t * 16 + colb] = (_Float16)gelu_exact(acc[t][r]);
    __syncthreads();

    for (int i = tid; i < 64 * 64; i += 128) {
        int k = i >> 6, n = i & 63;
        wbuf[n * 64 + k] = (_Float16)w2[i];
    }
    __syncthreads();

    float* resf = (float*)A1;   // reuse own A1 region (16x64 f32 = 4 KB)
    f16x16 a2f[2];
    a2f[0] = frag_a(A2, 64, 0, lane);
    a2f[1] = frag_a(A2, 64, 32, lane);
#pragma unroll
    for (int t = 0; t < 4; ++t) {
        float bias = b2[t * 16 + colb];
#pragma unroll
        for (int r = 0; r < 8; ++r) acc[t][r] = bias;
    }
#pragma unroll
    for (int kc = 0; kc < 2; ++kc)
#pragma unroll
        for (int t = 0; t < 4; ++t)
            acc[t] = wmma_f16(a2f[kc], frag_b(wbuf, 64, kc * 32, t * 16, lane), acc[t]);
#pragma unroll
    for (int t = 0; t < 4; ++t)
#pragma unroll
        for (int r = 0; r < 8; ++r) {
            int n = nbase + rowb + r; if (n >= N) n = N - 1;
            resf[(rowb + r) * 64 + t * 16 + colb] =
                acc[t][r] + xf[(unsigned)n * 64u + t * 16 + colb];  // residual
        }

    // layernorm: lanes 0..15 each own one row (LDS ops of a wave are in-order)
    if (lane < 16) {
        int n = nbase + lane;
        if (n < N) {
            const float* row = resf + lane * 64;
            float mu = 0.0f;
#pragma unroll
            for (int c = 0; c < 64; ++c) mu += row[c];
            mu *= (1.0f / 64.0f);
            float var = 0.0f;
#pragma unroll
            for (int c = 0; c < 64; ++c) { float d0 = row[c] - mu; var += d0 * d0; }
            var *= (1.0f / 64.0f);
            float rinv = rsqrtf(var + 1e-5f);
#pragma unroll
            for (int c = 0; c < 64; ++c)
                out[(unsigned)n * 64u + c] = (row[c] - mu) * rinv * lng[c] + lnb[c];
        }
    }
}

// ---------------------------------------------------------------------------
// Host launcher
// ---------------------------------------------------------------------------
static inline size_t align256(size_t x) { return (x + 255) & ~(size_t)255; }

extern "C" void kernel_launch(void* const* d_in, const int* in_sizes, int n_in,
                              void* d_out, int out_size, void* d_ws, size_t ws_size,
                              hipStream_t stream) {
    const float* xf   = (const float*)d_in[0];   // node_features [N,64]
    const int*   ei   = (const int*)d_in[1];     // edge_index [2,E]
    const float* ea   = (const float*)d_in[2];   // edge_attr [E,64]
    const float* mw1  = (const float*)d_in[3];
    const float* mb1  = (const float*)d_in[4];
    const float* mw2  = (const float*)d_in[5];
    const float* mb2  = (const float*)d_in[6];
    const float* qw   = (const float*)d_in[7];
    const float* qb   = (const float*)d_in[8];
    const float* kvw  = (const float*)d_in[9];
    const float* kvb  = (const float*)d_in[10];
    const float* ow1  = (const float*)d_in[11];
    const float* ob1  = (const float*)d_in[12];
    const float* ow2  = (const float*)d_in[13];
    const float* ob2  = (const float*)d_in[14];
    const float* lng  = (const float*)d_in[15];
    const float* lnb  = (const float*)d_in[16];
    float* out = (float*)d_out;

    const int N = in_sizes[0] / 64;
    const int E = in_sizes[1] / 2;

    // workspace carve-up
    char* w = (char*)d_ws;
    float* qn    = (float*)w;  w += align256((size_t)N * 64 * 4);
    float* mbuf  = (float*)w;  w += align256((size_t)E * 64 * 4);
    float* attn  = (float*)w;  w += align256((size_t)E * 8 * 4);
    int*   lastE = (int*)w;    w += align256((size_t)N * 4);
    float* pbuf  = (float*)w;  w += align256((size_t)N * 8 * 4);
    float* denom = (float*)w;  w += align256(8 * 4);
    float* agg   = (float*)w;  w += align256((size_t)N * 64 * 4);
    (void)ws_size; (void)n_in; (void)out_size;

    gc_init<<<((size_t)N * 64 + 255) / 256, 256, 0, stream>>>(agg, lastE, denom, N);
    gc_qn<<<(N + 63) / 64, 128, 0, stream>>>(xf, qw, qb, qn, N);
    gc_edge<<<(E + 63) / 64, 128, 0, stream>>>(xf, ei, ea, mw1, mb1, mw2, mb2,
                                               kvw, kvb, qn, mbuf, attn, lastE, N, E);
    gc_prep<<<((size_t)N * 8 + 255) / 256, 256, 0, stream>>>(attn, lastE, pbuf, denom, N);
    gc_scatter<<<((size_t)E * 8 + 255) / 256, 256, 0, stream>>>(mbuf, ei, pbuf, denom, agg, E);
    gc_out<<<(N + 63) / 64, 128, 0, stream>>>(xf, agg, ow1, ob1, ow2, ob2,
                                              lng, lnb, out, N);
}
